// MurelCell_5849745457858
// MI455X (gfx1250) — matrix operand order repaired and checked by reading, not compiled, not run on hardware
//
#include <hip/hip_runtime.h>
#include <hip/hip_bf16.h>

typedef __attribute__((ext_vector_type(16))) __bf16 v16bf;
typedef __attribute__((ext_vector_type(8)))  float  v8f;
typedef unsigned short u16;
typedef unsigned int   u32;

#define DEV __device__ __forceinline__

// ---------- bf16 helpers (RNE) ----------
DEV float bf2f(u16 u) {
    u32 v = ((u32)u) << 16;
    return __builtin_bit_cast(float, v);
}
DEV u16 f2bf(float f) {
    u32 u = __builtin_bit_cast(u32, f);
    u32 r = (u + 0x7FFFu + ((u >> 16) & 1u)) >> 16;
    return (u16)r;
}

union ABFrag { v16bf v; uint4 q[2]; };

// A fragment (16x32 bf16, wave32): lane half h reads K chunks [k0+h*8, +8) and [k0+16+h*8, +8)
DEV v16bf load_a_frag(const u16* rowPtr, int k0, int half) {
    ABFrag t;
    t.q[0] = *reinterpret_cast<const uint4*>(rowPtr + k0 + half * 8);
    t.q[1] = *reinterpret_cast<const uint4*>(rowPtr + k0 + 16 + half * 8);
    return t.v;
}
// B fragment from transposed weight (N x Kp): lane half h reads K chunks [k0+h*8, +8) and [k0+16+h*8, +8)
DEV v16bf load_b_frag(const u16* rowPtr, int k0, int half) {
    ABFrag t;
    t.q[0] = *reinterpret_cast<const uint4*>(rowPtr + k0 + half * 8);
    t.q[1] = *reinterpret_cast<const uint4*>(rowPtr + k0 + 16 + half * 8);
    return t.v;
}

DEV v8f wmma_bf16(v16bf a, v16bf b, v8f c) {
    return __builtin_amdgcn_wmma_f32_16x16x32_bf16(false, a, false, b, (short)0, c, false, false);
}

// ---------- problem constants ----------
#define RB    1152     // B*N rows
#define HH    1200
#define HP    1216     // H padded to multiple of 32
#define DOBJ  2048
#define DQ    4800
#define DOUT  2048
#define NN    36
#define BKc   64       // K chunk for relmax LDS staging
#define LDSW  72       // padded LDS row stride (halfwords) to break bank conflicts

// ---------- utility kernels ----------
__global__ void k_fill_u32(u32* p, int n) {
    int i = blockIdx.x * blockDim.x + threadIdx.x;
    if (i < n) p[i] = 0u;
}

__global__ void k_cast_bf16(const float* __restrict__ in, u16* __restrict__ out, int n) {
    int i = blockIdx.x * blockDim.x + threadIdx.x;
    if (i < n) out[i] = f2bf(in[i]);
}

// in: (K x N) f32 row-major  ->  out: (N x Kp) bf16, zero-padded K tail
__global__ void k_transpose_cast(const float* __restrict__ in, u16* __restrict__ out,
                                 int K, int N, int Kp) {
    int i = blockIdx.x * blockDim.x + threadIdx.x;
    int total = N * Kp;
    if (i >= total) return;
    int n = i / Kp, k = i - n * Kp;
    out[i] = (k < K) ? f2bf(in[(size_t)k * N + n]) : (u16)0;
}

// g0 = relu(box @ Wb0), g1 = relu(box @ Wb1), stored bf16 (RB x HP, padded)
__global__ void k_g01(const float* __restrict__ box, const float* __restrict__ Wb0,
                      const float* __restrict__ Wb1, u16* __restrict__ g0, u16* __restrict__ g1) {
    int i = blockIdx.x * blockDim.x + threadIdx.x;
    const int total = RB * HP;
    if (i >= total) return;
    int r = i / HP, h = i - r * HP;
    if (h < HH) {
        float b0 = box[r * 4 + 0], b1 = box[r * 4 + 1], b2 = box[r * 4 + 2], b3 = box[r * 4 + 3];
        float s0 = b0 * Wb0[0 * HH + h] + b1 * Wb0[1 * HH + h] + b2 * Wb0[2 * HH + h] + b3 * Wb0[3 * HH + h];
        float s1 = b0 * Wb1[0 * HH + h] + b1 * Wb1[1 * HH + h] + b2 * Wb1[2 * HH + h] + b3 * Wb1[3 * HH + h];
        g0[i] = f2bf(fmaxf(s0, 0.f));
        g1[i] = f2bf(fmaxf(s1, 0.f));
    } else {
        g0[i] = 0; g1[i] = 0;
    }
}

// ---------- generic bf16 WMMA GEMM: C(MxN) = A(MxK) @ Bt(NxK)^T ----------
// EPI 1: outF = relu(acc)
// EPI 2: outB = bf16(relu(acc) * aux)        (bilinear fusion product)
// EPI 3: outF = acc, outB = bf16(acc)        (fused: keep both precisions)
// EPI 4: outB = bf16(relu(acc))
template <int EPI>
__global__ void k_gemm(const u16* __restrict__ A, const u16* __restrict__ Bt,
                       int M, int N, int K, int lda, int ldb,
                       float* __restrict__ outF, int ldo,
                       u16* __restrict__ outB, int ldob,
                       const float* __restrict__ aux, int ldaux) {
    int wid  = (blockIdx.x * blockDim.x + threadIdx.x) >> 5;
    int lane = threadIdx.x & 31;
    int tilesN = N >> 4;
    int tm = wid / tilesN, tn = wid - tm * tilesN;
    if (tm >= (M >> 4)) return;
    int lr = lane & 15, half = lane >> 4;
    const u16* aRow = A  + (size_t)(tm * 16 + lr) * lda;
    const u16* bRow = Bt + (size_t)(tn * 16 + lr) * ldb;
    v8f acc = {0.f, 0.f, 0.f, 0.f, 0.f, 0.f, 0.f, 0.f};
    for (int k = 0; k < K; k += 32) {
        v16bf a = load_a_frag(aRow, k, half);
        v16bf b = load_b_frag(bRow, k, half);
        acc = wmma_bf16(a, b, acc);
    }
#pragma unroll
    for (int rr = 0; rr < 8; ++rr) {
        int gm = tm * 16 + half * 8 + rr;
        int gn = tn * 16 + lr;
        float v = acc[rr];
        if constexpr (EPI == 1) {
            outF[(size_t)gm * ldo + gn] = fmaxf(v, 0.f);
        } else if constexpr (EPI == 2) {
            float p = fmaxf(v, 0.f) * aux[(size_t)gm * ldaux + gn];
            outB[(size_t)gm * ldob + gn] = f2bf(p);
        } else if constexpr (EPI == 3) {
            outF[(size_t)gm * ldo + gn] = v;
            outB[(size_t)gm * ldob + gn] = f2bf(v);
        } else {
            outB[(size_t)gm * ldob + gn] = f2bf(fmaxf(v, 0.f));
        }
    }
}

// ---------- dominant kernel: pairwise relations + max over j, fused epilogue ----------
// grid: 1152 (b,i) rows x 16 O-tiles of 128; block: 256 threads (8 waves, one 16-col slice each)
// Double-buffered LDS product tiles: one barrier per K-chunk; the next chunk's VALU/global
// product build co-executes with the current chunk's WMMA stream.
__global__ void k_relmax(const u16* __restrict__ bH0, const u16* __restrict__ bH1,
                         const u16* __restrict__ bG0, const u16* __restrict__ bG1,
                         const u16* __restrict__ bWpoT, const u16* __restrict__ bWboT,
                         const float* __restrict__ obj, const float* __restrict__ fused,
                         float* __restrict__ out) {
    __shared__ __align__(16) u16 sPf[2][48 * LDSW];
    __shared__ __align__(16) u16 sPb[2][48 * LDSW];

    int blk = blockIdx.x;
    int r   = blk >> 4;    // 0..1151  (= b*36 + i)
    int ot  = blk & 15;
    int b   = r / NN;
    int rowBase = b * NN;

    int tid  = threadIdx.x;
    int wave = tid >> 5, lane = tid & 31;
    int lr = lane & 15, half = lane >> 4;
    int obase = ot * 128 + wave * 16;

    const u16* wpRow  = bWpoT + (size_t)(obase + lr) * HP;
    const u16* wbRow  = bWboT + (size_t)(obase + lr) * HP;
    const u16* h0Row  = bH0 + (size_t)r * HP;
    const u16* g0Row  = bG0 + (size_t)r * HP;
    const u16* h1Base = bH1 + (size_t)rowBase * HP;
    const u16* g1Base = bG1 + (size_t)rowBase * HP;

    auto fill = [&](int buf, int kc) {
        for (int e = tid; e < 48 * BKc; e += 256) {
            int jr = e / BKc, kk = e - jr * BKc;
            int gk = kc + kk;
            int di = jr * LDSW + kk;
            if (jr < NN) {
                sPf[buf][di] = f2bf(bf2f(h0Row[gk]) * bf2f(h1Base[(size_t)jr * HP + gk]));
                sPb[buf][di] = f2bf(bf2f(g0Row[gk]) * bf2f(g1Base[(size_t)jr * HP + gk]));
            } else {
                sPf[buf][di] = 0;
                sPb[buf][di] = 0;
            }
        }
    };

    v8f acc0 = {0.f, 0.f, 0.f, 0.f, 0.f, 0.f, 0.f, 0.f};
    v8f acc1 = {0.f, 0.f, 0.f, 0.f, 0.f, 0.f, 0.f, 0.f};
    v8f acc2 = {0.f, 0.f, 0.f, 0.f, 0.f, 0.f, 0.f, 0.f};

    const int NCHUNK = HP / BKc;  // 19
    fill(0, 0);
    for (int c = 0; c < NCHUNK; ++c) {
        int kc = c * BKc;
        __syncthreads();                                   // fill of buf (c&1) complete
        if (c + 1 < NCHUNK) fill((c + 1) & 1, kc + BKc);   // overlaps with WMMAs below
        const u16* pf = sPf[c & 1];
        const u16* pb = sPb[c & 1];
#pragma unroll
        for (int ks = 0; ks < 2; ++ks) {
            int k0g = kc + ks * 32;
            int klo = ks * 32;
            // preload all fragments, then issue 6 back-to-back WMMAs
            v16bf bp = load_b_frag(wpRow, k0g, half);
            v16bf bb = load_b_frag(wbRow, k0g, half);
            v16bf a0 = load_a_frag(pf + (size_t)(0 + lr) * LDSW, klo, half);
            v16bf a1 = load_a_frag(pf + (size_t)(16 + lr) * LDSW, klo, half);
            v16bf a2 = load_a_frag(pf + (size_t)(32 + lr) * LDSW, klo, half);
            v16bf a3 = load_a_frag(pb + (size_t)(0 + lr) * LDSW, klo, half);
            v16bf a4 = load_a_frag(pb + (size_t)(16 + lr) * LDSW, klo, half);
            v16bf a5 = load_a_frag(pb + (size_t)(32 + lr) * LDSW, klo, half);
            // interleave accumulators: dependent chains are 3 WMMAs apart
            acc0 = wmma_bf16(a0, bp, acc0);
            acc1 = wmma_bf16(a1, bp, acc1);
            acc2 = wmma_bf16(a2, bp, acc2);
            acc0 = wmma_bf16(a3, bb, acc0);
            acc1 = wmma_bf16(a4, bb, acc1);
            acc2 = wmma_bf16(a5, bb, acc2);
        }
    }

    // max over valid j rows. lane holds rows m = half*8 + rr of each 16-row j-tile, col n = lr.
    float m = -3.402823466e38f;
#pragma unroll
    for (int rr = 0; rr < 8; ++rr) m = fmaxf(m, fmaxf(acc0[rr], acc1[rr]));
    if (half == 0) {
#pragma unroll
        for (int rr = 0; rr < 4; ++rr) m = fmaxf(m, acc2[rr]);  // j = 32..35 only
    }
    m = fmaxf(m, __shfl_xor(m, 16, 32));  // combine the two half-wave row groups
    if (half == 0) {
        size_t gi = (size_t)r * DOUT + obase + lr;
        out[gi] = obj[gi] + fused[gi] + m;
    }
}

// ---------- host launcher ----------
extern "C" void kernel_launch(void* const* d_in, const int* in_sizes, int n_in,
                              void* d_out, int out_size, void* d_ws, size_t ws_size,
                              hipStream_t stream) {
    (void)in_sizes; (void)n_in; (void)out_size; (void)ws_size;

    const float* q   = (const float*)d_in[0];
    const float* obj = (const float*)d_in[1];
    const float* box = (const float*)d_in[2];
    const float* Wf0 = (const float*)d_in[3];
    const float* Wf1 = (const float*)d_in[4];
    const float* Wfo = (const float*)d_in[5];
    const float* Wb0 = (const float*)d_in[6];
    const float* Wb1 = (const float*)d_in[7];
    const float* Wbo = (const float*)d_in[8];
    const float* Wp0 = (const float*)d_in[9];
    const float* Wp1 = (const float*)d_in[10];
    const float* Wpo = (const float*)d_in[11];
    float* out = (float*)d_out;

    char* ws = (char*)d_ws;
    size_t cur = 0;
    auto alloc = [&](size_t bytes) -> char* {
        char* p = ws + cur;
        cur = (cur + bytes + 255) & ~(size_t)255;
        return p;
    };

    u16*   bWpoT  = (u16*)alloc((size_t)DOUT * HP * 2);
    u16*   bWboT  = (u16*)alloc((size_t)DOUT * HP * 2);
    u16*   bH0    = (u16*)alloc((size_t)RB * HP * 2);
    u16*   bH1    = (u16*)alloc((size_t)RB * HP * 2);
    u16*   bG0    = (u16*)alloc((size_t)RB * HP * 2);
    u16*   bG1    = (u16*)alloc((size_t)RB * HP * 2);
    float* fFused = (float*)alloc((size_t)RB * DOUT * 4);
    u16*   bFused = (u16*)alloc((size_t)RB * DOUT * 2);
    u16*   bWf0T  = (u16*)alloc((size_t)HH * DOBJ * 2);
    u16*   bWf1T  = (u16*)alloc((size_t)HH * DQ * 2);
    u16*   bWfoT  = (u16*)alloc((size_t)DOUT * HP * 2);
    u16*   bWp0T  = (u16*)alloc((size_t)HH * DOBJ * 2);
    u16*   bWp1T  = (u16*)alloc((size_t)HH * DOBJ * 2);
    u16*   bObj   = (u16*)alloc((size_t)RB * DOBJ * 2);
    u16*   bQ     = (u16*)alloc((size_t)RB * DQ * 2);
    float* t0     = (float*)alloc((size_t)RB * HH * 4);
    u16*   bProd  = (u16*)alloc((size_t)RB * HP * 2);

    const int TPB = 256;
    auto nb = [](int n) { return (n + 255) / 256; };

    // zero bf16 buffers whose K-pad tail is only partially written by GEMM epilogues
    {
        int nu = RB * HP / 2;  // u32 count per buffer
        k_fill_u32<<<nb(nu), TPB, 0, stream>>>((u32*)bProd, nu);
        k_fill_u32<<<nb(nu), TPB, 0, stream>>>((u32*)bH0, nu);
        k_fill_u32<<<nb(nu), TPB, 0, stream>>>((u32*)bH1, nu);
    }

    // casts
    k_cast_bf16<<<nb(RB * DOBJ), TPB, 0, stream>>>(obj, bObj, RB * DOBJ);
    k_cast_bf16<<<nb(RB * DQ), TPB, 0, stream>>>(q, bQ, RB * DQ);

    // weight transposes (store as N x Kp, bf16)
    k_transpose_cast<<<nb(HH * DOBJ), TPB, 0, stream>>>(Wf0, bWf0T, DOBJ, HH, DOBJ);
    k_transpose_cast<<<nb(HH * DQ), TPB, 0, stream>>>(Wf1, bWf1T, DQ, HH, DQ);
    k_transpose_cast<<<nb(DOUT * HP), TPB, 0, stream>>>(Wfo, bWfoT, HH, DOUT, HP);
    k_transpose_cast<<<nb(HH * DOBJ), TPB, 0, stream>>>(Wp0, bWp0T, DOBJ, HH, DOBJ);
    k_transpose_cast<<<nb(HH * DOBJ), TPB, 0, stream>>>(Wp1, bWp1T, DOBJ, HH, DOBJ);
    k_transpose_cast<<<nb(DOUT * HP), TPB, 0, stream>>>(Wpo, bWpoT, HH, DOUT, HP);
    k_transpose_cast<<<nb(DOUT * HP), TPB, 0, stream>>>(Wbo, bWboT, HH, DOUT, HP);

    // box projections (tiny K=4) on VALU
    k_g01<<<nb(RB * HP), TPB, 0, stream>>>(box, Wb0, Wb1, bG0, bG1);

    auto gblocks = [](int M, int N) { return ((M / 16) * (N / 16) * 32 + 255) / 256; };

    // GEMM1: t0 = relu(obj @ Wf0)                (1152x1200, K=2048)
    k_gemm<1><<<gblocks(RB, HH), TPB, 0, stream>>>(bObj, bWf0T, RB, HH, DOBJ, DOBJ, DOBJ,
                                                   t0, HH, nullptr, 0, nullptr, 0);
    // GEMM2: bProd = bf16(relu(q @ Wf1) * t0)    (1152x1200, K=4800)
    k_gemm<2><<<gblocks(RB, HH), TPB, 0, stream>>>(bQ, bWf1T, RB, HH, DQ, DQ, DQ,
                                                   nullptr, 0, bProd, HP, t0, HH);
    // GEMM3: fused = bProd @ Wfo                 (1152x2048, K=1216 padded)
    k_gemm<3><<<gblocks(RB, DOUT), TPB, 0, stream>>>(bProd, bWfoT, RB, DOUT, HP, HP, HP,
                                                     fFused, DOUT, bFused, DOUT, nullptr, 0);
    // GEMM4/5: h0 = relu(fused @ Wp0), h1 = relu(fused @ Wp1)
    k_gemm<4><<<gblocks(RB, HH), TPB, 0, stream>>>(bFused, bWp0T, RB, HH, DOBJ, DOBJ, DOBJ,
                                                   nullptr, 0, bH0, HP, nullptr, 0);
    k_gemm<4><<<gblocks(RB, HH), TPB, 0, stream>>>(bFused, bWp1T, RB, HH, DOBJ, DOBJ, DOBJ,
                                                   nullptr, 0, bH1, HP, nullptr, 0);

    // dominant stage: pairwise relations + max over j + fused final add
    k_relmax<<<RB * 16, TPB, 0, stream>>>(bH0, bH1, bG0, bG1, bWpoT, bWboT, obj, fFused, out);
}
